// Transcoders_58360015618337
// MI455X (gfx1250) — matrix operand order repaired and checked by reading, not compiled, not run on hardware
//
#include <hip/hip_runtime.h>
#include <cstdint>
#include <cstddef>

// Problem dimensions (fixed by the reference)
#define B_ROWS 4096
#define D_IN   1024
#define D_LAT  16384

typedef __attribute__((ext_vector_type(16))) __bf16          v16bf;
typedef __attribute__((ext_vector_type(16))) unsigned short  v16us;
typedef __attribute__((ext_vector_type(8)))  unsigned short  us8;
typedef __attribute__((ext_vector_type(8)))  float           v8f;

// ---------------- bf16 helpers (RNE, manual integer path) ----------------------------
__device__ __forceinline__ unsigned short f2bf(float f) {
  unsigned u = __float_as_uint(f);
  u += 0x7FFFu + ((u >> 16) & 1u);          // round-to-nearest-even
  return (unsigned short)(u >> 16);
}
__device__ __forceinline__ float bf2f(unsigned short h) {
  return __uint_as_float(((unsigned)h) << 16);
}
__device__ __forceinline__ void f1_hl(float f, unsigned short& h, unsigned short& l) {
  unsigned short hb = f2bf(f);
  h = hb;
  l = f2bf(f - bf2f(hb));                   // residual for hi/lo split (fp32-like accuracy)
}

// Monotone key for radix-select over floats (larger float -> larger key)
__device__ __forceinline__ unsigned f2key(float f) {
  unsigned u = __float_as_uint(f);
  return (u & 0x80000000u) ? ~u : (u | 0x80000000u);
}
__device__ __forceinline__ float key2f(unsigned k) {
  unsigned u = (k & 0x80000000u) ? (k ^ 0x80000000u) : ~k;
  return __uint_as_float(u);
}

// Load a 16-element bf16 fragment as two 8-element chunks `gap` elements apart.
// A fragment: base = row + hi*8,  gap = 16   (K = hi*8+[0,8) and 16+hi*8+[0,8))
// B fragment: base = row + hi*8,  gap = 16   (K = hi*8+[0,8) and 16+hi*8+[0,8))
__device__ __forceinline__ v16bf ldfrag2(const unsigned short* __restrict__ p, int gap) {
  const us8 a = *(const us8*)p;
  const us8 b = *(const us8*)(p + gap);
  v16us u = __builtin_shufflevector(a, b, 0, 1, 2, 3, 4, 5, 6, 7,
                                          8, 9, 10, 11, 12, 13, 14, 15);
  return __builtin_bit_cast(v16bf, u);
}

// =====================================================================================
// Conversion kernels: fp32 -> (hi, lo) bf16 planes, done ONCE per operand so the GEMM
// inner loop carries zero conversion VALU work.
// =====================================================================================
__global__ __launch_bounds__(256) void cvt_hl_kernel(
    const float* __restrict__ in, unsigned short* __restrict__ h,
    unsigned short* __restrict__ l, int n /* multiple of 4 */)
{
  for (int i = (blockIdx.x * 256 + threadIdx.x) * 4; i < n; i += gridDim.x * 256 * 4) {
    const float4 f = *(const float4*)(in + i);
    unsigned short hx, hy, hz, hw, lx, ly, lz, lw;
    f1_hl(f.x, hx, lx); f1_hl(f.y, hy, ly);
    f1_hl(f.z, hz, lz); f1_hl(f.w, hw, lw);
    uint2 hp, lp;
    hp.x = ((unsigned)hy << 16) | hx;  hp.y = ((unsigned)hw << 16) | hz;
    lp.x = ((unsigned)ly << 16) | lx;  lp.y = ((unsigned)lw << 16) | lz;
    *(uint2*)(h + i) = hp;
    *(uint2*)(l + i) = lp;
  }
}

// Transpose-convert W_skip [K][N] -> bf16 hi/lo planes in [N][K] layout so the skip
// GEMM uses the same contiguous NT path as the encoder GEMM.
__global__ __launch_bounds__(256) void cvtT_hl_kernel(
    const float* __restrict__ in, unsigned short* __restrict__ outh,
    unsigned short* __restrict__ outl)
{
  __shared__ float tile[32][33];
  const int tx = threadIdx.x & 31;
  const int ty = threadIdx.x >> 5;   // 8 rows of loaders
  const int k0 = blockIdx.x * 32;
  const int n0 = blockIdx.y * 32;
  for (int r = ty; r < 32; r += 8)
    tile[r][tx] = in[(size_t)(k0 + r) * D_IN + n0 + tx];
  __syncthreads();
  for (int r = ty; r < 32; r += 8) {
    unsigned short h, l;
    f1_hl(tile[tx][r], h, l);
    outh[(size_t)(n0 + r) * D_IN + k0 + tx] = h;
    outl[(size_t)(n0 + r) * D_IN + k0 + tx] = l;
  }
}

// =====================================================================================
// Generic NT GEMM on pre-converted bf16 hi/lo planes:
//   out[M][N] = A[M][K] * B[N][K]^T + bias[N]
// bf16 hi/lo split: 3x v_wmma_f32_16x16x32_bf16 per K-step per tile.
// Workgroup = 8 waves (4 M-slices x 2 N-slices); wave tile = 16(M) x 128(N).
// All loads use constant immediate offsets (t*32KB, +16/+32B) from FOUR persistent
// base pointers bumped by 64B per K-step -> no per-load address VALU, so no
// WMMA->VALU WAR hazard nops on fragment registers. N-tiles processed in pairs so
// two independent accumulator chains interleave.
// =====================================================================================
__global__ __launch_bounds__(256) void gemm_nt_bf16hl_kernel(
    const unsigned short* __restrict__ Ah, const unsigned short* __restrict__ Al,
    const unsigned short* __restrict__ Bh, const unsigned short* __restrict__ Bl,
    const float* __restrict__ bias, float* __restrict__ out, int N)
{
  const int lane = threadIdx.x & 31;
  const int wave = threadIdx.x >> 5;
  const int lm   = lane & 15;
  const int hi   = lane >> 4;
  const int m0   = blockIdx.y * 64 + (wave & 3) * 16;
  const int n0   = blockIdx.x * 256 + (wave >> 2) * 128;

  // persistent base pointers; all per-load offsets are compile-time immediates
  const unsigned short* pAh = Ah + (size_t)(m0 + lm) * D_IN + hi * 8;
  const unsigned short* pAl = Al + (size_t)(m0 + lm) * D_IN + hi * 8;
  const unsigned short* pBh = Bh + (size_t)(n0 + lm) * D_IN + hi * 8;
  const unsigned short* pBl = Bl + (size_t)(n0 + lm) * D_IN + hi * 8;

  v8f acc[8];
#pragma unroll
  for (int t = 0; t < 8; ++t)
    acc[t] = (v8f){0.f, 0.f, 0.f, 0.f, 0.f, 0.f, 0.f, 0.f};

  for (int k0 = 0; k0 < D_IN; k0 += 32) {
    const v16bf fAh = ldfrag2(pAh, 16);
    const v16bf fAl = ldfrag2(pAl, 16);
#pragma unroll
    for (int t = 0; t < 8; t += 2) {
      const int r0 = t * 16 * D_IN;          // 32768*t bytes: fits 24-bit IOFFSET
      const int r1 = (t + 1) * 16 * D_IN;
      const v16bf fBh0 = ldfrag2(pBh + r0, 16);
      const v16bf fBl0 = ldfrag2(pBl + r0, 16);
      const v16bf fBh1 = ldfrag2(pBh + r1, 16);
      const v16bf fBl1 = ldfrag2(pBl + r1, 16);
      // interleave the two independent accumulator chains
      acc[t]     = __builtin_amdgcn_wmma_f32_16x16x32_bf16(false, fAh, false, fBh0, (short)0, acc[t],     false, false);
      acc[t + 1] = __builtin_amdgcn_wmma_f32_16x16x32_bf16(false, fAh, false, fBh1, (short)0, acc[t + 1], false, false);
      acc[t]     = __builtin_amdgcn_wmma_f32_16x16x32_bf16(false, fAh, false, fBl0, (short)0, acc[t],     false, false);
      acc[t + 1] = __builtin_amdgcn_wmma_f32_16x16x32_bf16(false, fAh, false, fBl1, (short)0, acc[t + 1], false, false);
      acc[t]     = __builtin_amdgcn_wmma_f32_16x16x32_bf16(false, fAl, false, fBh0, (short)0, acc[t],     false, false);
      acc[t + 1] = __builtin_amdgcn_wmma_f32_16x16x32_bf16(false, fAl, false, fBh1, (short)0, acc[t + 1], false, false);
    }
    pAh += 32; pAl += 32; pBh += 32; pBl += 32;   // 64-byte bumps of dedicated addr regs
  }
  // C/D layout: lane = (m>=8)*16 + n ; VGPR r = row (m&7)
#pragma unroll
  for (int t = 0; t < 8; ++t) {
    const int n_g = n0 + t * 16 + lm;
    const float b = bias[n_g];
    float* op = out + (size_t)(m0 + hi * 8) * N + n_g;
#pragma unroll
    for (int r = 0; r < 8; ++r) op[(size_t)r * N] = acc[t][r] + b;
  }
}

// =====================================================================================
// Per-column sum / sum-of-squares of x (deterministic, no float atomics)
// total_variance = sum_j (Q_j - S_j^2 / B)
// =====================================================================================
__global__ __launch_bounds__(256) void colstats_kernel(
    const float* __restrict__ x, float* __restrict__ colS, float* __restrict__ colQ)
{
  __shared__ float sS[16][17];
  __shared__ float sQ[16][17];
  const int tx  = threadIdx.x & 15;
  const int ty  = threadIdx.x >> 4;
  const int col = blockIdx.x * 16 + tx;
  float S = 0.f, Q = 0.f;
  for (int r = ty; r < B_ROWS; r += 16) {
    const float v = x[(size_t)r * D_IN + col];
    S += v; Q += v * v;
  }
  sS[ty][tx] = S; sQ[ty][tx] = Q;
  __syncthreads();
  if (ty == 0) {
    float s = 0.f, q = 0.f;
#pragma unroll
    for (int k = 0; k < 16; ++k) { s += sS[k][tx]; q += sQ[k][tx]; }
    colS[col] = s; colQ[col] = q;
  }
}

__global__ void init_kernel(unsigned* __restrict__ l0cnt)
{
  if (threadIdx.x == 0) *l0cnt = 0u;
}

// =====================================================================================
// Per-row work. One WG (256 thr) per row; pre row (64 KB) held in LDS (320 KB/WGP).
//  - exact 8-bit x 4-pass radix-select for the k-th largest value, k = 32/128/256
//  - mean_encoded row = relu(v) * mult/3 ; l0 count for k=256
//  - staged sparse decode (nested top-k sets): rank-1 updates gathered from
//    L2-resident W_dec (64 MB < 192 MB L2); loss per stage; mean_decoded row
// All float reductions fixed-order (deterministic); compaction via ballot prefix.
// =====================================================================================
#define ROW_SMEM_BYTES 69760  // 65536 row + 1024 hist + 1024 lidx + 1024 lval + 1024 red + 128 scal

__global__ __launch_bounds__(256) void row_kernel(
    const float* __restrict__ pre, const float* __restrict__ skip,
    const float* __restrict__ mlp, const float* __restrict__ Wdec,
    float* __restrict__ meanEnc, float* __restrict__ meanDec,
    float* __restrict__ lossRow, unsigned* __restrict__ l0cnt)
{
  extern __shared__ unsigned char smem_raw[];
  float*    row  = (float*)smem_raw;                        // 16384 f
  unsigned* hist = (unsigned*)(smem_raw + 65536);           // 256 u
  unsigned* lidx = (unsigned*)(smem_raw + 66560);           // 256 u
  float*    lval = (float*)(smem_raw + 67584);              // 256 f
  float*    red  = (float*)(smem_raw + 68608);              // 256 f
  unsigned* scal = (unsigned*)(smem_raw + 69632);           // 32 u

  const int tid  = threadIdx.x;
  const int lane = tid & 31;
  const int wv   = tid >> 5;
  const int i    = blockIdx.x;
  const float* prow = pre + (size_t)i * D_LAT;

  for (int j = tid; j < D_LAT; j += 256) row[j] = prow[j];
  if (tid == 0) scal[2] = 0u;  // block l0 accumulator
  __syncthreads();

  // ---- radix select: exact k-th largest per row for k in {32,128,256} ----
  float thr[3];
  const int kv[3] = {32, 128, 256};
  for (int s = 0; s < 3; ++s) {
    unsigned prefix = 0;
    unsigned kk = (unsigned)kv[s];
    for (int pass = 0; pass < 4; ++pass) {
      const int shb = 24 - 8 * pass;
      hist[tid] = 0u;
      __syncthreads();
      for (int j = tid; j < D_LAT; j += 256) {
        const unsigned key = f2key(row[j]);
        const bool match = (pass == 0) || ((key >> (shb + 8)) == prefix);
        if (match) atomicAdd(&hist[(key >> shb) & 255u], 1u);
      }
      __syncthreads();
      if (tid == 0) {
        unsigned cum = 0;
        int d = 255;
        for (; d > 0; --d) {
          if (cum + hist[d] >= kk) break;
          cum += hist[d];
        }
        scal[0] = (unsigned)d;
        scal[1] = kk - cum;
      }
      __syncthreads();
      prefix = (prefix << 8) | scal[0];
      kk = scal[1];
      __syncthreads();
    }
    thr[s] = key2f(prefix);
  }
  const float t32 = thr[0], t128 = thr[1], t256 = thr[2];

  // ---- mean_encoded + l0 (multiplicity: 3 if >=t32, 2 if >=t128, 1 if >=t256) ----
  unsigned l0loc = 0;
  float* meRow = meanEnc + (size_t)i * D_LAT;
  for (int j = tid; j < D_LAT; j += 256) {
    const float v = row[j];
    const int m = (v >= t32) ? 3 : (v >= t128) ? 2 : (v >= t256) ? 1 : 0;
    const float rv = v > 0.f ? v : 0.f;
    meRow[j] = rv * (float)m * (1.f / 3.f);
    if (m != 0 && v > 0.f) ++l0loc;
  }
  atomicAdd(&scal[2], l0loc);  // integer LDS atomic (deterministic)

  // ---- staged sparse decode; each thread owns 4 output columns in registers ----
  const int c0 = tid * 4;
  const float4 sk = *(const float4*)(skip + (size_t)i * D_IN + c0);
  const float4 mo = *(const float4*)(mlp  + (size_t)i * D_IN + c0);
  float ax = 0.f, ay = 0.f, az = 0.f, aw = 0.f;
  float mdx = 0.f, mdy = 0.f, mdz = 0.f, mdw = 0.f;
  float lossLoc = 0.f;
  const float INFV = __uint_as_float(0x7f800000u);

  for (int s = 0; s < 3; ++s) {
    const float lo  = thr[s];
    const float hiT = (s == 0) ? INFV : thr[s - 1];
    for (int base = 0; base < D_LAT; base += 256) {
      const int j = base + tid;
      const float v = row[j];
      const bool p = (v >= lo) && (v < hiT) && (v > 0.f);
      const unsigned m32 = __builtin_amdgcn_ballot_w32(p);
      if (lane == 0) scal[8 + wv] = (unsigned)__popc(m32);
      __syncthreads();
      if (tid == 0) {
        unsigned ssum = 0;
#pragma unroll
        for (int w = 0; w < 8; ++w) { scal[16 + w] = ssum; ssum += scal[8 + w]; }
        scal[3] = ssum;
      }
      __syncthreads();
      if (p) {
        const unsigned slot = scal[16 + wv] + (unsigned)__popc(m32 & ((1u << lane) - 1u));
        lidx[slot] = (unsigned)j;
        lval[slot] = v;
      }
      __syncthreads();
      const unsigned n = scal[3];
      for (unsigned e = 0; e < n; ++e) {
        const float val = lval[e];
        const float4 w = *(const float4*)(Wdec + (size_t)lidx[e] * D_IN + c0);
        ax += val * w.x; ay += val * w.y; az += val * w.z; aw += val * w.w;
      }
      __syncthreads();
    }
    const float dx = ax + sk.x, dy = ay + sk.y, dz = az + sk.z, dw = aw + sk.w;
    mdx += dx; mdy += dy; mdz += dz; mdw += dw;
    const float ex = mo.x - dx, ey = mo.y - dy, ez = mo.z - dz, ew = mo.w - dw;
    lossLoc += ex * ex + ey * ey + ez * ez + ew * ew;
  }
  {
    float4 md;
    md.x = mdx * (1.f / 3.f); md.y = mdy * (1.f / 3.f);
    md.z = mdz * (1.f / 3.f); md.w = mdw * (1.f / 3.f);
    *(float4*)(meanDec + (size_t)i * D_IN + c0) = md;
  }
  // deterministic loss reduction
  red[tid] = lossLoc;
  __syncthreads();
  for (int off = 128; off > 0; off >>= 1) {
    if (tid < off) red[tid] += red[tid + off];
    __syncthreads();
  }
  if (tid == 0) {
    lossRow[i] = red[0];
    atomicAdd(l0cnt, scal[2]);
  }
}

// =====================================================================================
// Scalars. tv = sum_j (Q_j - S_j^2/B); loss = sum_i lossRow[i] / tv; l0.
// =====================================================================================
__global__ __launch_bounds__(256) void finalize_kernel(
    const float* __restrict__ colS, const float* __restrict__ colQ,
    const float* __restrict__ lossRow, const unsigned* __restrict__ l0cnt,
    float* __restrict__ outScalars)
{
  __shared__ float red[256];
  const int tid = threadIdx.x;
  float tv = 0.f;
  for (int c = tid; c < D_IN; c += 256) {
    const float s = colS[c];
    tv += colQ[c] - s * s * (1.f / (float)B_ROWS);
  }
  red[tid] = tv;
  __syncthreads();
  for (int off = 128; off > 0; off >>= 1) {
    if (tid < off) red[tid] += red[tid + off];
    __syncthreads();
  }
  const float tvTot = red[0];
  __syncthreads();
  float ls = 0.f;
  for (int r = tid; r < B_ROWS; r += 256) ls += lossRow[r];
  red[tid] = ls;
  __syncthreads();
  for (int off = 128; off > 0; off >>= 1) {
    if (tid < off) red[tid] += red[tid + off];
    __syncthreads();
  }
  if (tid == 0) {
    outScalars[0] = red[0] / tvTot;
    outScalars[1] = (float)(*l0cnt);
  }
}

// =====================================================================================
// Host launcher
// =====================================================================================
extern "C" void kernel_launch(void* const* d_in, const int* in_sizes, int n_in,
                              void* d_out, int out_size, void* d_ws, size_t ws_size,
                              hipStream_t stream)
{
  (void)in_sizes; (void)n_in; (void)out_size; (void)ws_size;

  const float* x     = (const float*)d_in[0];
  const float* mlp   = (const float*)d_in[1];
  const float* Wenc  = (const float*)d_in[2];
  const float* benc  = (const float*)d_in[3];
  const float* Wdec  = (const float*)d_in[4];
  const float* bdec  = (const float*)d_in[5];
  const float* Wskip = (const float*)d_in[6];

  float* out = (float*)d_out;
  float* meanEnc = out;                                   // 4096*16384
  float* meanDec = out + (size_t)B_ROWS * D_LAT;          // 4096*1024
  float* scalars = meanDec + (size_t)B_ROWS * D_IN;       // [loss, l0]

  // workspace layout (~374 MB):
  //   pre | skip | colS | colQ | lossRow | l0(+pad) | xh | xl | weh | wel | wsth | wstl
  float* ws      = (float*)d_ws;
  float* pre     = ws;                                    // 4096*16384 f
  float* skip    = pre + (size_t)B_ROWS * D_LAT;          // 4096*1024 f
  float* colS    = skip + (size_t)B_ROWS * D_IN;          // 1024 f
  float* colQ    = colS + D_IN;                           // 1024 f
  float* lossRow = colQ + D_IN;                           // 4096 f
  unsigned* l0cnt = (unsigned*)(lossRow + B_ROWS);        // 1 u (+15 pad)
  unsigned short* xh   = (unsigned short*)(lossRow + B_ROWS + 16);
  unsigned short* xl   = xh   + (size_t)B_ROWS * D_IN;    // 4096*1024 us each
  unsigned short* weh  = xl   + (size_t)B_ROWS * D_IN;
  unsigned short* wel  = weh  + (size_t)D_LAT * D_IN;     // 16384*1024 us each
  unsigned short* wsth = wel  + (size_t)D_LAT * D_IN;
  unsigned short* wstl = wsth + (size_t)D_IN * D_IN;      // 1024*1024 us each

  (void)hipFuncSetAttribute(reinterpret_cast<const void*>(row_kernel),
                            hipFuncAttributeMaxDynamicSharedMemorySize,
                            (int)ROW_SMEM_BYTES);

  init_kernel<<<1, 64, 0, stream>>>(l0cnt);
  colstats_kernel<<<D_IN / 16, 256, 0, stream>>>(x, colS, colQ);

  // one-shot fp32 -> bf16 hi/lo conversion of all GEMM operands
  cvt_hl_kernel<<<(B_ROWS * D_IN) / (256 * 4), 256, 0, stream>>>(x, xh, xl, B_ROWS * D_IN);
  cvt_hl_kernel<<<(D_LAT * D_IN) / (256 * 4), 256, 0, stream>>>(Wenc, weh, wel, D_LAT * D_IN);
  cvtT_hl_kernel<<<dim3(D_IN / 32, D_IN / 32), 256, 0, stream>>>(Wskip, wsth, wstl);

  // pre  = x @ Wenc^T + benc   (M=4096, N=16384, K=1024)
  gemm_nt_bf16hl_kernel<<<dim3(D_LAT / 256, B_ROWS / 64), 256, 0, stream>>>(
      xh, xl, weh, wel, benc, pre, D_LAT);
  // skip = x @ Wskip + bdec    (via transposed-converted Wskip; M=4096, N=1024, K=1024)
  gemm_nt_bf16hl_kernel<<<dim3(D_IN / 256, B_ROWS / 64), 256, 0, stream>>>(
      xh, xl, wsth, wstl, bdec, skip, D_IN);

  row_kernel<<<B_ROWS, 256, ROW_SMEM_BYTES, stream>>>(pre, skip, mlp, Wdec,
                                                      meanEnc, meanDec, lossRow, l0cnt);
  finalize_kernel<<<1, 256, 0, stream>>>(colS, colQ, lossRow, l0cnt, scalars);
}
